// LSTMMultiHead_10866267259163
// MI455X (gfx1250) — matrix operand order repaired and checked by reading, not compiled, not run on hardware
//
#include <hip/hip_runtime.h>
#include <hip/hip_bf16.h>
#include <math.h>

#define HID   256
#define FOURH 1024
#define BATCH 256
#define AGENTS 20
#define TLEN  50
#define INFEAT 6
#define NNBR  19
#define NHEAD 8
#define HDIM  32
#define DSTEPS 60
#define NBR_N (BATCH*NNBR)   /* 4864 */

typedef __attribute__((ext_vector_type(16))) _Float16 v16h;
typedef __attribute__((ext_vector_type(8)))  float    v8f;

union U16h { v16h v; _Float16 h[16]; };

__device__ __forceinline__ float sigmoidf_(float x){ return 1.0f/(1.0f+__expf(-x)); }

// Load the per-lane A operand (16x32 f16 tile) from an f32 row, converting to f16.
// Layout (wave32): lane<16 -> row=lo, halves 0..15 = k kb+0..15
//                  lane>=16 -> row=lo, halves 0..15 = kb+16..31
__device__ __forceinline__ v16h loadA16(const float* __restrict__ rowp, int kb, int hi){
  U16h a;
  const float4* p0 = (const float4*)(rowp + kb + hi*16);
  const float4* p1 = (const float4*)(rowp + kb + hi*16 + 8);
  float4 f0=p0[0], f1=p0[1], f2=p1[0], f3=p1[1];
  a.h[0]=(_Float16)f0.x; a.h[1]=(_Float16)f0.y; a.h[2]=(_Float16)f0.z; a.h[3]=(_Float16)f0.w;
  a.h[4]=(_Float16)f1.x; a.h[5]=(_Float16)f1.y; a.h[6]=(_Float16)f1.z; a.h[7]=(_Float16)f1.w;
  a.h[8]=(_Float16)f2.x; a.h[9]=(_Float16)f2.y; a.h[10]=(_Float16)f2.z; a.h[11]=(_Float16)f2.w;
  a.h[12]=(_Float16)f3.x; a.h[13]=(_Float16)f3.y; a.h[14]=(_Float16)f3.z; a.h[15]=(_Float16)f3.w;
  return a.v;
}

// ---------------- prep: weight swizzle into per-lane WMMA B-operand layout ----------------
// dest index: [ntile][kblk][lane][h] ; lane = (s%16) + (k%32>=16 ? 16:0), h = k%16
// gatePerm: n = gate*256+hc  ->  s = (hc/16)*64 + gate*16 + (hc%16)   (gate-interleaved tiles)
__global__ void swizzle_w(const float* __restrict__ srcA, const float* __restrict__ srcB,
                          int Ksplit, int K, int N, int gatePerm, int convMode,
                          _Float16* __restrict__ dst)
{
  int KB = K >> 5;
  int tot = N*K;
  for (int idx = blockIdx.x*blockDim.x + threadIdx.x; idx < tot; idx += gridDim.x*blockDim.x) {
    int n = idx / K, k = idx - n*K;
    float val;
    if (convMode) {
      int c  = (k < 512) ? k : (k-512);
      int kk = (k < 512) ? 0 : 1;
      val = srcA[((size_t)n*512 + c)*3 + kk];
    } else if (k < Ksplit) {
      val = srcA[(size_t)n*Ksplit + k];
    } else {
      val = srcB[(size_t)n*(K-Ksplit) + (k-Ksplit)];
    }
    int s = gatePerm ? (((n&255)>>4)*64 + ((n>>8)<<4) + (n&15)) : n;
    int ntile = s>>4, lo = s&15;
    int kblk = k>>5;
    int lane = lo + (((k&31)>=16)?16:0);
    int hh = k & 15;
    dst[(((size_t)ntile*KB + kblk)*32 + lane)*16 + hh] = (_Float16)val;
  }
}

__global__ void prep_bias(const float* __restrict__ bih, const float* __restrict__ bhh,
                          float* __restrict__ out){
  int n = blockIdx.x*blockDim.x + threadIdx.x;
  if (n >= FOURH) return;
  int s = ((n&255)>>4)*64 + ((n>>8)<<4) + (n&15);
  out[s] = bih[n]+bhh[n];
}

__global__ void prep_wx(const float* __restrict__ W, float* __restrict__ out){
  int n = blockIdx.x*blockDim.x + threadIdx.x;
  if (n >= FOURH) return;
  int s = ((n&255)>>4)*64 + ((n>>8)<<4) + (n&15);
  for (int d=0; d<INFEAT; ++d) out[s*8+d] = W[n*INFEAT+d];
  out[s*8+6]=0.0f; out[s*8+7]=0.0f;
}

// ---------------- generic WMMA GEMM: C[M,N] = A[M,K] @ W^T (+bias), wave tile 16x64 ---------
__global__ void gemm_ws(int M, int N, int K, const float* __restrict__ A, int lda,
                        const _Float16* __restrict__ W, const float* __restrict__ bias,
                        float* __restrict__ C, int ldc)
{
  int wid  = blockIdx.x*(blockDim.x>>5) + (threadIdx.x>>5);
  int lane = threadIdx.x & 31;
  int ng = N >> 6;
  int total = (M>>4)*ng;
  if (wid >= total) return;
  int mtile = wid / ng, ntile0 = (wid - mtile*ng)*4;
  int lo = lane & 15, hi = lane >> 4;

  v8f acc[4];
  for (int g=0; g<4; ++g) {
    float b0 = bias ? bias[(ntile0+g)*16 + lo] : 0.0f;
    for (int r=0; r<8; ++r) acc[g][r] = b0;
  }
  int KB = K>>5;
  const float* arow = A + (size_t)(mtile*16 + lo)*lda;
  const v16h* Wp = (const v16h*)W;
  for (int kb=0; kb<K; kb+=32) {
    v16h av = loadA16(arow, kb, hi);
    int kblk = kb>>5;
    for (int g=0; g<4; ++g) {
      U16h b; b.v = Wp[((size_t)(ntile0+g)*KB + kblk)*32 + lane];
      acc[g] = __builtin_amdgcn_wmma_f32_16x16x32_f16(false, av, false, b.v,
                                                      (short)0, acc[g], false, false);
    }
  }
  for (int g=0; g<4; ++g) {
    int n = (ntile0+g)*16 + lo;
    for (int r=0; r<8; ++r) {
      int m = mtile*16 + r + hi*8;
      C[(size_t)m*ldc + n] = acc[g][r];
    }
  }
}

// ---------------- fused LSTM step: gates = init + A0@W0^T (+A1@W1^T) ; cell update --------
// W is gate-interleaved swizzled over combined K = K0+K1 (columns 4H).
// mode 0: init = preGate[m][s] (swizzled order, biases already included)
// mode 1: init = biasSwz[s] + x_t[m,0:6] @ Wx  (layer-0 input projection, K=6)
// mode 2: init = biasSwz[s]
__global__ void lstm_step(int M,
   const float* __restrict__ A0, int K0,
   const float* __restrict__ A1, int K1,
   const _Float16* __restrict__ W,
   const float* __restrict__ preGate,
   const float* __restrict__ biasSwz,
   const float* __restrict__ xSrc, const float* __restrict__ Wx, int t, int nbrMode,
   int mode,
   float* __restrict__ cbuf, float* __restrict__ hout)
{
  int wid  = blockIdx.x*(blockDim.x>>5) + (threadIdx.x>>5);
  int lane = threadIdx.x & 31;
  int total = (M>>4)*16;           // H/16 = 16 hidden blocks
  if (wid >= total) return;
  int mtile = wid>>4, hblk = wid&15;
  int lo = lane&15, hi = lane>>4;

  v8f acc[4];
  if (mode == 0) {
    for (int g=0; g<4; ++g) {
      int s = hblk*64 + g*16 + lo;
      for (int r=0; r<8; ++r)
        acc[g][r] = preGate[(size_t)(mtile*16 + r + hi*8)*FOURH + s];
    }
  } else {
    float bz[4]; float wxv[4][INFEAT];
    for (int g=0; g<4; ++g){
      int s = hblk*64 + g*16 + lo;
      bz[g] = biasSwz[s];
      if (mode==1) for (int d=0; d<INFEAT; ++d) wxv[g][d] = Wx[s*8+d];
    }
    for (int g=0; g<4; ++g) for (int r=0; r<8; ++r) acc[g][r]=bz[g];
    if (mode==1) {
      for (int r=0; r<8; ++r) {
        int m = mtile*16 + r + hi*8;
        const float* xp;
        if (nbrMode) { int b = m/NNBR, j = m - b*NNBR;
                       xp = xSrc + (((size_t)b*AGENTS + 1 + j)*TLEN + t)*INFEAT; }
        else         { xp = xSrc + (((size_t)m*AGENTS)*TLEN + t)*INFEAT; }
        float xv[INFEAT];
        for (int d=0; d<INFEAT; ++d) xv[d]=xp[d];
        for (int g=0; g<4; ++g){
          float sacc=0.0f;
          for (int d=0; d<INFEAT; ++d) sacc += xv[d]*wxv[g][d];
          acc[g][r]+=sacc;
        }
      }
    }
  }

  int KBt = (K0+K1)>>5;
  const v16h* Wp = (const v16h*)W;
  int row = mtile*16 + lo;

  const float* a0 = A0 + (size_t)row*K0;
  for (int kb=0; kb<K0; kb+=32) {
    v16h av = loadA16(a0, kb, hi);
    int kblk = kb>>5;
    for (int g=0; g<4; ++g) {
      U16h b; b.v = Wp[((size_t)(hblk*4+g)*KBt + kblk)*32 + lane];
      acc[g] = __builtin_amdgcn_wmma_f32_16x16x32_f16(false, av, false, b.v,
                                                      (short)0, acc[g], false, false);
    }
  }
  if (K1 > 0) {
    const float* a1 = A1 + (size_t)row*K1;
    for (int kb=0; kb<K1; kb+=32) {
      v16h av = loadA16(a1, kb, hi);
      int kblk = (K0+kb)>>5;
      for (int g=0; g<4; ++g) {
        U16h b; b.v = Wp[((size_t)(hblk*4+g)*KBt + kblk)*32 + lane];
        acc[g] = __builtin_amdgcn_wmma_f32_16x16x32_f16(false, av, false, b.v,
                                                        (short)0, acc[g], false, false);
      }
    }
  }

  // fused cell update: PyTorch gate order i,f,g,o
  int p = hblk*16 + lo;
  for (int r=0; r<8; ++r){
    int m = mtile*16 + r + hi*8;
    float i_=acc[0][r], f_=acc[1][r], g_=acc[2][r], o_=acc[3][r];
    float cp = cbuf[(size_t)m*HID + p];
    float c  = sigmoidf_(f_)*cp + sigmoidf_(i_)*tanhf(g_);
    float hv = sigmoidf_(o_)*tanhf(c);
    cbuf[(size_t)m*HID + p] = c;
    hout[(size_t)m*HID + p] = hv;
  }
}

// ---------------- attention over neighbors, only at t in {T-2, T-1} ----------------------
__global__ void attn_kernel(const float* __restrict__ q, const float* __restrict__ k,
                            const float* __restrict__ v, float* __restrict__ out)
{
  int wid  = blockIdx.x*(blockDim.x>>5) + (threadIdx.x>>5);
  int lane = threadIdx.x & 31;
  int total = 2*BATCH*NHEAD;
  if (wid >= total) return;
  int tt = wid / (BATCH*NHEAD);
  int rem = wid - tt*(BATCH*NHEAD);
  int b = rem / NHEAD, hd = rem - b*NHEAD;
  int qrow = tt*BATCH + b;
  float qv = q[(size_t)qrow*HID + hd*HDIM + lane];
  const float scale = 0.17677669529663687f;  // 1/sqrt(32)
  float sc[NNBR];
  for (int j=0; j<NNBR; ++j){
    int krow = tt*NBR_N + b*NNBR + j;
    float p = qv * k[(size_t)krow*HID + hd*HDIM + lane];
    for (int off=16; off>0; off>>=1) p += __shfl_xor(p, off, 32);
    sc[j] = p*scale;
  }
  float mx = sc[0];
  for (int j=1; j<NNBR; ++j) mx = fmaxf(mx, sc[j]);
  float sum=0.0f;
  for (int j=0; j<NNBR; ++j){ sc[j]=__expf(sc[j]-mx); sum+=sc[j]; }
  float inv = 1.0f/sum;
  float o=0.0f;
  for (int j=0; j<NNBR; ++j){
    int vrow = tt*NBR_N + b*NNBR + j;
    o += sc[j]*inv * v[(size_t)vrow*HID + hd*HDIM + lane];
  }
  out[(size_t)qrow*HID + hd*HDIM + lane] = o;
}

// ---------------- gather conv input rows: [ego(T-2), att(T-2), ego(T-1), att(T-1)] ------
__global__ void gather_comb(const float* __restrict__ egoSel, const float* __restrict__ attP,
                            float* __restrict__ cA){
  int idx = blockIdx.x*blockDim.x + threadIdx.x;
  if (idx >= BATCH*1024) return;
  int b = idx >> 10, c = idx & 1023;
  float val;
  if      (c < 256) val = egoSel[(size_t)b*HID + c];
  else if (c < 512) val = attP  [(size_t)b*HID + (c-256)];
  else if (c < 768) val = egoSel[(size_t)(BATCH + b)*HID + (c-512)];
  else              val = attP  [(size_t)(BATCH + b)*HID + (c-768)];
  cA[idx] = val;
}

// ---------------- final projection to 2 outputs, transposed to [B, DT, 2] ---------------
__global__ void out_proj(const float* __restrict__ decH1, const float* __restrict__ Wout,
                         const float* __restrict__ bout, float* __restrict__ out){
  int idx = blockIdx.x*blockDim.x + threadIdx.x;
  if (idx >= BATCH*DSTEPS) return;
  int b = idx / DSTEPS, t = idx - b*DSTEPS;
  const float* hrow = decH1 + ((size_t)t*BATCH + b)*HID;
  for (int kk=0; kk<2; ++kk){
    float acc = bout[kk];
    const float* wr = Wout + kk*HID;
    for (int i=0; i<HID; ++i) acc += hrow[i]*wr[i];
    out[(size_t)idx*2 + kk] = acc;
  }
}

// =========================================================================================
extern "C" void kernel_launch(void* const* d_in, const int* in_sizes, int n_in,
                              void* d_out, int out_size, void* d_ws, size_t ws_size,
                              hipStream_t stream)
{
  (void)in_sizes; (void)n_in; (void)out_size; (void)ws_size;
  const float* x        = (const float*)d_in[0];
  const float* egoWih0  = (const float*)d_in[1];
  const float* egoWhh0  = (const float*)d_in[2];
  const float* egoBih0  = (const float*)d_in[3];
  const float* egoBhh0  = (const float*)d_in[4];
  const float* egoWih1  = (const float*)d_in[5];
  const float* egoWhh1  = (const float*)d_in[6];
  const float* egoBih1  = (const float*)d_in[7];
  const float* egoBhh1  = (const float*)d_in[8];
  const float* nbrWih0  = (const float*)d_in[9];
  const float* nbrWhh0  = (const float*)d_in[10];
  const float* nbrBih0  = (const float*)d_in[11];
  const float* nbrBhh0  = (const float*)d_in[12];
  const float* nbrWih1  = (const float*)d_in[13];
  const float* nbrWhh1  = (const float*)d_in[14];
  const float* nbrBih1  = (const float*)d_in[15];
  const float* nbrBhh1  = (const float*)d_in[16];
  const float* decWih0  = (const float*)d_in[17];
  const float* decWhh0  = (const float*)d_in[18];
  const float* decBih0  = (const float*)d_in[19];
  const float* decBhh0  = (const float*)d_in[20];
  const float* decWih1  = (const float*)d_in[21];
  const float* decWhh1  = (const float*)d_in[22];
  const float* decBih1  = (const float*)d_in[23];
  const float* decBhh1  = (const float*)d_in[24];
  const float* Wq       = (const float*)d_in[25];
  const float* Wk       = (const float*)d_in[26];
  const float* Wv       = (const float*)d_in[27];
  const float* Wao      = (const float*)d_in[28];
  const float* bq       = (const float*)d_in[29];
  const float* bk       = (const float*)d_in[30];
  const float* bv       = (const float*)d_in[31];
  const float* bao      = (const float*)d_in[32];
  const float* Wconv    = (const float*)d_in[33];
  const float* bconv    = (const float*)d_in[34];
  const float* Wout     = (const float*)d_in[35];
  const float* bout     = (const float*)d_in[36];

  // --- bump allocator over workspace ---
  char* wsb = (char*)d_ws; size_t off = 0;
  auto alloc = [&](size_t bytes)->void* {
    void* p = wsb + off; off += (bytes + 255) & ~(size_t)255; return p;
  };

  // f16 swizzled weights
  _Float16* egoWhh0s = (_Float16*)alloc((size_t)FOURH*HID*2);
  _Float16* nbrWhh0s = (_Float16*)alloc((size_t)FOURH*HID*2);
  _Float16* decWhh0s = (_Float16*)alloc((size_t)FOURH*HID*2);
  _Float16* egoL1s   = (_Float16*)alloc((size_t)FOURH*512*2);
  _Float16* nbrL1s   = (_Float16*)alloc((size_t)FOURH*512*2);
  _Float16* decL1s   = (_Float16*)alloc((size_t)FOURH*512*2);
  _Float16* Wqs      = (_Float16*)alloc((size_t)HID*HID*2);
  _Float16* Wks      = (_Float16*)alloc((size_t)HID*HID*2);
  _Float16* Wvs      = (_Float16*)alloc((size_t)HID*HID*2);
  _Float16* Waos     = (_Float16*)alloc((size_t)HID*HID*2);
  _Float16* Wconvs   = (_Float16*)alloc((size_t)HID*1024*2);
  _Float16* decWih0s = (_Float16*)alloc((size_t)FOURH*HID*2);
  // f32 small prep
  float* egoWxs = (float*)alloc((size_t)FOURH*8*4);
  float* nbrWxs = (float*)alloc((size_t)FOURH*8*4);
  float* egoB0s = (float*)alloc(FOURH*4);
  float* egoB1s = (float*)alloc(FOURH*4);
  float* nbrB0s = (float*)alloc(FOURH*4);
  float* nbrB1s = (float*)alloc(FOURH*4);
  float* decB0s = (float*)alloc(FOURH*4);
  float* decB1s = (float*)alloc(FOURH*4);
  // activations
  size_t nbrF = (size_t)NBR_N*HID, egoF = (size_t)BATCH*HID;
  float* zbuf    = (float*)alloc(nbrF*4);
  float* nbr_h0a = (float*)alloc(nbrF*4);
  float* nbr_h0b = (float*)alloc(nbrF*4);
  float* nbr_c0  = (float*)alloc(nbrF*4);
  float* nbr_h1a = (float*)alloc(nbrF*4);
  float* nbr_h1b = (float*)alloc(nbrF*4);
  float* nbr_c1  = (float*)alloc(nbrF*4);
  float* nbrSel  = (float*)alloc(2*nbrF*4);
  float* ego_h0a = (float*)alloc(egoF*4);
  float* ego_h0b = (float*)alloc(egoF*4);
  float* ego_c0  = (float*)alloc(egoF*4);
  float* ego_h1a = (float*)alloc(egoF*4);
  float* ego_h1b = (float*)alloc(egoF*4);
  float* ego_c1  = (float*)alloc(egoF*4);
  float* egoSel  = (float*)alloc(2*egoF*4);
  float* qbuf    = (float*)alloc(2*egoF*4);
  float* kbuf    = (float*)alloc(2*nbrF*4);
  float* vbuf    = (float*)alloc(2*nbrF*4);
  float* attO    = (float*)alloc(2*egoF*4);
  float* attP    = (float*)alloc(2*egoF*4);
  float* cA      = (float*)alloc((size_t)BATCH*1024*4);
  float* tfLast  = (float*)alloc(egoF*4);
  float* preDec0 = (float*)alloc((size_t)BATCH*FOURH*4);
  float* dec_h0a = (float*)alloc(egoF*4);
  float* dec_h0b = (float*)alloc(egoF*4);
  float* dec_c0  = (float*)alloc(egoF*4);
  float* dec_c1  = (float*)alloc(egoF*4);
  float* decH1   = (float*)alloc((size_t)DSTEPS*BATCH*HID*4);

  // --- one-time weight prep (cheap; re-run every call for determinism) ---
  swizzle_w<<<512,256,0,stream>>>(egoWhh0, nullptr, HID, HID, FOURH, 1, 0, egoWhh0s);
  swizzle_w<<<512,256,0,stream>>>(nbrWhh0, nullptr, HID, HID, FOURH, 1, 0, nbrWhh0s);
  swizzle_w<<<512,256,0,stream>>>(decWhh0, nullptr, HID, HID, FOURH, 1, 0, decWhh0s);
  swizzle_w<<<512,256,0,stream>>>(egoWih1, egoWhh1, HID, 512, FOURH, 1, 0, egoL1s);
  swizzle_w<<<512,256,0,stream>>>(nbrWih1, nbrWhh1, HID, 512, FOURH, 1, 0, nbrL1s);
  swizzle_w<<<512,256,0,stream>>>(decWih1, decWhh1, HID, 512, FOURH, 1, 0, decL1s);
  swizzle_w<<<256,256,0,stream>>>(Wq,  nullptr, HID, HID, HID, 0, 0, Wqs);
  swizzle_w<<<256,256,0,stream>>>(Wk,  nullptr, HID, HID, HID, 0, 0, Wks);
  swizzle_w<<<256,256,0,stream>>>(Wv,  nullptr, HID, HID, HID, 0, 0, Wvs);
  swizzle_w<<<256,256,0,stream>>>(Wao, nullptr, HID, HID, HID, 0, 0, Waos);
  swizzle_w<<<512,256,0,stream>>>(Wconv, nullptr, 1024, 1024, HID, 0, 1, Wconvs);
  swizzle_w<<<512,256,0,stream>>>(decWih0, nullptr, HID, HID, FOURH, 1, 0, decWih0s);
  prep_bias<<<4,256,0,stream>>>(egoBih0, egoBhh0, egoB0s);
  prep_bias<<<4,256,0,stream>>>(egoBih1, egoBhh1, egoB1s);
  prep_bias<<<4,256,0,stream>>>(nbrBih0, nbrBhh0, nbrB0s);
  prep_bias<<<4,256,0,stream>>>(nbrBih1, nbrBhh1, nbrB1s);
  prep_bias<<<4,256,0,stream>>>(decBih0, decBhh0, decB0s);
  prep_bias<<<4,256,0,stream>>>(decBih1, decBhh1, decB1s);
  prep_wx<<<4,256,0,stream>>>(egoWih0, egoWxs);
  prep_wx<<<4,256,0,stream>>>(nbrWih0, nbrWxs);

  // --- zero initial states ---
  hipMemsetAsync(zbuf,   0, nbrF*4, stream);
  hipMemsetAsync(nbr_c0, 0, nbrF*4, stream);
  hipMemsetAsync(nbr_c1, 0, nbrF*4, stream);
  hipMemsetAsync(ego_c0, 0, egoF*4, stream);
  hipMemsetAsync(ego_c1, 0, egoF*4, stream);
  hipMemsetAsync(dec_c0, 0, egoF*4, stream);
  hipMemsetAsync(dec_c1, 0, egoF*4, stream);

  auto stepBlocks = [](int M){ return ((M>>4)*16 + 7)/8; };
  auto gemmBlocks = [](int M, int N){ return ((M>>4)*(N>>6) + 7)/8; };

  // --- neighbor encoder (2-layer LSTM over T=50) ---
  {
    const float* h0p = zbuf; const float* h1p = zbuf;
    for (int t=0; t<TLEN; ++t){
      float* h0o = (t&1)? nbr_h0b : nbr_h0a;
      lstm_step<<<stepBlocks(NBR_N),256,0,stream>>>(NBR_N, h0p, HID, nullptr, 0, nbrWhh0s,
          nullptr, nbrB0s, x, nbrWxs, t, 1, 1, nbr_c0, h0o);
      float* h1o;
      if      (t == TLEN-2) h1o = nbrSel;
      else if (t == TLEN-1) h1o = nbrSel + nbrF;
      else h1o = (t&1)? nbr_h1b : nbr_h1a;
      lstm_step<<<stepBlocks(NBR_N),256,0,stream>>>(NBR_N, h0o, HID, h1p, HID, nbrL1s,
          nullptr, nbrB1s, nullptr, nullptr, 0, 0, 2, nbr_c1, h1o);
      h0p = h0o; h1p = h1o;
    }
  }
  // --- ego encoder ---
  {
    const float* h0p = zbuf; const float* h1p = zbuf;
    for (int t=0; t<TLEN; ++t){
      float* h0o = (t&1)? ego_h0b : ego_h0a;
      lstm_step<<<stepBlocks(BATCH),256,0,stream>>>(BATCH, h0p, HID, nullptr, 0, egoWhh0s,
          nullptr, egoB0s, x, egoWxs, t, 0, 1, ego_c0, h0o);
      float* h1o;
      if      (t == TLEN-2) h1o = egoSel;
      else if (t == TLEN-1) h1o = egoSel + egoF;
      else h1o = (t&1)? ego_h1b : ego_h1a;
      lstm_step<<<stepBlocks(BATCH),256,0,stream>>>(BATCH, h0o, HID, h1p, HID, egoL1s,
          nullptr, egoB1s, nullptr, nullptr, 0, 0, 2, ego_c1, h1o);
      h0p = h0o; h1p = h1o;
    }
  }

  // --- attention path (only t = T-2, T-1 feed the conv tap used by the decoder) ---
  gemm_ws<<<gemmBlocks(2*BATCH,HID),256,0,stream>>>(2*BATCH, HID, HID, egoSel, HID, Wqs, bq, qbuf, HID);
  gemm_ws<<<gemmBlocks(2*NBR_N,HID),256,0,stream>>>(2*NBR_N, HID, HID, nbrSel, HID, Wks, bk, kbuf, HID);
  gemm_ws<<<gemmBlocks(2*NBR_N,HID),256,0,stream>>>(2*NBR_N, HID, HID, nbrSel, HID, Wvs, bv, vbuf, HID);
  attn_kernel<<<(2*BATCH*NHEAD+7)/8,256,0,stream>>>(qbuf, kbuf, vbuf, attO);
  gemm_ws<<<gemmBlocks(2*BATCH,HID),256,0,stream>>>(2*BATCH, HID, HID, attO, HID, Waos, bao, attP, HID);

  // --- conv tap at t = T-1 (kernel width 3, pad 1: uses comb[T-2], comb[T-1]) ---
  gather_comb<<<(BATCH*1024+255)/256,256,0,stream>>>(egoSel, attP, cA);
  gemm_ws<<<gemmBlocks(BATCH,HID),256,0,stream>>>(BATCH, HID, 1024, cA, 1024, Wconvs, bconv, tfLast, HID);

  // --- decoder: constant input -> precompute layer-0 input gates once ---
  gemm_ws<<<gemmBlocks(BATCH,FOURH),256,0,stream>>>(BATCH, FOURH, HID, tfLast, HID,
                                                    decWih0s, decB0s, preDec0, FOURH);
  {
    const float* h0p = zbuf; const float* h1p = zbuf;
    for (int t=0; t<DSTEPS; ++t){
      float* h0o = (t&1)? dec_h0b : dec_h0a;
      lstm_step<<<stepBlocks(BATCH),256,0,stream>>>(BATCH, h0p, HID, nullptr, 0, decWhh0s,
          preDec0, nullptr, nullptr, nullptr, 0, 0, 0, dec_c0, h0o);
      float* h1o = decH1 + (size_t)t*BATCH*HID;
      lstm_step<<<stepBlocks(BATCH),256,0,stream>>>(BATCH, h0o, HID, h1p, HID, decL1s,
          nullptr, decB1s, nullptr, nullptr, 0, 0, 2, dec_c1, h1o);
      h0p = h0o; h1p = h1o;
    }
  }
  out_proj<<<(BATCH*DSTEPS+255)/256,256,0,stream>>>(decH1, Wout, bout, (float*)d_out);
}
